// SelfAttention_42030549959218
// MI455X (gfx1250) — matrix operand order repaired and checked by reading, not compiled, not run on hardware
//
#include <hip/hip_runtime.h>

// ---------------------------------------------------------------------------
// Self-attention for MI455X (gfx1250): f16 WMMA (16x16x32) with fp32 accum.
// B=2, S=2048, D=1024, H=16, Dh=64.
// GEMMs: 128x128 block tile, 32x64 wave tile (8 WMMA / 6 fragment loads).
// Attention: flash-style online softmax, 64-key blocks (8+8 WMMA per block).
// ---------------------------------------------------------------------------

typedef __attribute__((ext_vector_type(16))) _Float16 v16h;
typedef __attribute__((ext_vector_type(8)))  _Float16 v8h;
typedef __attribute__((ext_vector_type(8)))  float    v8f;

#define WMMA_F16(A_, B_, C_) \
  __builtin_amdgcn_wmma_f32_16x16x32_f16(false, (A_), false, (B_), (short)0, (C_), false, false)

#define ATT_B   2
#define ATT_S   2048
#define ATT_D   1024
#define ATT_H   16
#define ATT_DH  64
#define ATT_M   (ATT_B * ATT_S)   // 4096 rows
#define ATT_3D  (3 * ATT_D)       // 3072

union V16HU { v16h v; v8h h[2]; };

// A fragment (16x32 f16): lane holds row m = lane&15; halves 0..7 = K(kg*8..),
// halves 8..15 = K(16+kg*8..) with kg = lane>>4.  p points at (m0,k0).
__device__ __forceinline__ v16h load_a_frag(const _Float16* __restrict__ p, int lda) {
  const int lane = threadIdx.x & 31;
  const int m  = lane & 15;
  const int kg = (lane >> 4) << 3;  // 0 or 8
  V16HU u;
  u.h[0] = *(const v8h*)(p + (size_t)m * lda + kg);
  u.h[1] = *(const v8h*)(p + (size_t)m * lda + 16 + kg);
  return u.v;
}

// B fragment (32x16 f16) from an N-major buffer Bt[n][k]: lane holds column
// n = lane&15; halves 0..7 = K(kg..), halves 8..15 = K(16+kg..), kg = (lane>>4)*8.
// p points at (k0, n0), i.e. Bt + n0*ldn + k0.
__device__ __forceinline__ v16h load_b_frag(const _Float16* __restrict__ p, int ldn) {
  const int lane = threadIdx.x & 31;
  const int n  = lane & 15;
  const int kb = (lane >> 4) << 3;  // 0 or 8
  V16HU u;
  u.h[0] = *(const v8h*)(p + (size_t)n * ldn + kb);
  u.h[1] = *(const v8h*)(p + (size_t)n * ldn + kb + 16);
  return u.v;
}

// ------------------------------- conversion --------------------------------
__global__ void cvt_f32_to_f16_kernel(const float* __restrict__ in,
                                      _Float16* __restrict__ out, int n) {
  int i = blockIdx.x * 256 + threadIdx.x;
  if (i < n) out[i] = (_Float16)in[i];
}

// in[K][N] (fp32, row-major) -> out[N][K] (f16, N-major)
__global__ void transpose_cvt_kernel(const float* __restrict__ in,
                                     _Float16* __restrict__ out, int K, int N) {
  int i = blockIdx.x * 256 + threadIdx.x;
  if (i < K * N) {
    int k = i / N, n = i - k * N;
    out[(size_t)n * K + k] = (_Float16)in[i];
  }
}

// ------------------------------- QKV GEMM ----------------------------------
// C[4096,3072] = Xh @ Wqkv + b_qkv, scattered into
//   Qb [b,h,s,dh] | Kb [b,h,s,dh] | Vt [b,h,dh,s]   (all f16)
// Block = 256 thr = 8 waves (4m x 2n); block tile 128x128; wave tile 32x64.
__global__ __launch_bounds__(256) void qkv_gemm_kernel(
    const _Float16* __restrict__ Xh,    // [4096,1024]
    const _Float16* __restrict__ Wt,    // [3072,1024]  (N-major)
    const float*    __restrict__ bias,  // [3072]
    _Float16* __restrict__ Qb, _Float16* __restrict__ Kb,
    _Float16* __restrict__ Vt) {
  const int lane = threadIdx.x & 31;
  const int wave = threadIdx.x >> 5;
  const int m0 = blockIdx.x * 128 + (wave >> 1) * 32;
  const int n0 = blockIdx.y * 128 + (wave & 1) * 64;

  const v8f zero = {0.f, 0.f, 0.f, 0.f, 0.f, 0.f, 0.f, 0.f};
  v8f acc[2][4];
#pragma unroll
  for (int im = 0; im < 2; ++im)
#pragma unroll
    for (int jn = 0; jn < 4; ++jn) acc[im][jn] = zero;

  for (int k0 = 0; k0 < ATT_D; k0 += 32) {
    v16h a0 = load_a_frag(Xh + (size_t)m0        * ATT_D + k0, ATT_D);
    v16h a1 = load_a_frag(Xh + (size_t)(m0 + 16) * ATT_D + k0, ATT_D);
    v16h b0 = load_b_frag(Wt + (size_t)(n0)      * ATT_D + k0, ATT_D);
    v16h b1 = load_b_frag(Wt + (size_t)(n0 + 16) * ATT_D + k0, ATT_D);
    v16h b2 = load_b_frag(Wt + (size_t)(n0 + 32) * ATT_D + k0, ATT_D);
    v16h b3 = load_b_frag(Wt + (size_t)(n0 + 48) * ATT_D + k0, ATT_D);
    acc[0][0] = WMMA_F16(a0, b0, acc[0][0]);
    acc[0][1] = WMMA_F16(a0, b1, acc[0][1]);
    acc[0][2] = WMMA_F16(a0, b2, acc[0][2]);
    acc[0][3] = WMMA_F16(a0, b3, acc[0][3]);
    acc[1][0] = WMMA_F16(a1, b0, acc[1][0]);
    acc[1][1] = WMMA_F16(a1, b1, acc[1][1]);
    acc[1][2] = WMMA_F16(a1, b2, acc[1][2]);
    acc[1][3] = WMMA_F16(a1, b3, acc[1][3]);
  }

  const int nlo = lane & 15;
  const int mb  = (lane >> 4) << 3;
#pragma unroll
  for (int im = 0; im < 2; ++im) {
#pragma unroll
    for (int jn = 0; jn < 4; ++jn) {
      const int n = n0 + jn * 16 + nlo;
      const int which = n >> 10;       // 0=Q, 1=K, 2=V
      const int dm = n & (ATT_D - 1);
      const int h = dm >> 6, dh = dm & 63;
      const float bv = bias[n];
#pragma unroll
      for (int r = 0; r < 8; ++r) {
        const int m = m0 + im * 16 + mb + r;
        const int b = m >> 11, s = m & (ATT_S - 1);
        const _Float16 hv = (_Float16)(acc[im][jn][r] + bv);
        const size_t bh = (size_t)(b * ATT_H + h);
        if (which == 0)      Qb[(bh * ATT_S + s) * ATT_DH + dh] = hv;
        else if (which == 1) Kb[(bh * ATT_S + s) * ATT_DH + dh] = hv;
        else                 Vt[(bh * ATT_DH + dh) * ATT_S + s] = hv;
      }
    }
  }
}

// ------------------------------- attention ---------------------------------
// Block = 128 thr = 4 independent waves; each wave owns 16 query rows.
// Grid = (S/64, B*H).  Key blocks of 64 (4 score fragments).
__global__ __launch_bounds__(128) void attn_kernel(
    const _Float16* __restrict__ Qb, const _Float16* __restrict__ Kb,
    const _Float16* __restrict__ Vt, _Float16* __restrict__ AO) {
  __shared__ __align__(16) _Float16 Plds[4][16][64];   // per-wave P tile (2KB)

  const int lane = threadIdx.x & 31;
  const int wave = threadIdx.x >> 5;
  const int bh = blockIdx.y;
  const int b = bh >> 4, h = bh & 15;
  const int q0 = blockIdx.x * 64 + wave * 16;

  const _Float16* Qp = Qb + ((size_t)bh * ATT_S + q0) * ATT_DH;
  const _Float16* Kp = Kb + (size_t)bh * ATT_S * ATT_DH;
  const _Float16* Vp = Vt + (size_t)bh * ATT_DH * ATT_S;

  const v16h qa0 = load_a_frag(Qp,      ATT_DH);   // d = 0..31
  const v16h qa1 = load_a_frag(Qp + 32, ATT_DH);   // d = 32..63

  const v8f zero = {0.f, 0.f, 0.f, 0.f, 0.f, 0.f, 0.f, 0.f};
  v8f o[4];  o[0] = zero; o[1] = zero; o[2] = zero; o[3] = zero;
  float mrow[8], lrow[8];
#pragma unroll
  for (int r = 0; r < 8; ++r) { mrow[r] = -3.0e38f; lrow[r] = 0.f; }

  const float scale = 0.125f;   // 1/sqrt(64)
  const int nlo = lane & 15;
  const int grp = lane >> 4;    // which 8-row half this lane's C frag holds

  for (int j0 = 0; j0 < ATT_S; j0 += 64) {
    // ---- scores: S[16x64] = Q(16x64) @ K^T, four 16-key C fragments -------
    v8f s[4];
#pragma unroll
    for (int c = 0; c < 4; ++c) {
      const _Float16* kp = Kp + (size_t)(j0 + c * 16) * ATT_DH;
      v16h kb0 = load_b_frag(kp,      ATT_DH);
      v16h kb1 = load_b_frag(kp + 32, ATT_DH);
      v8f sc = zero;
      sc = WMMA_F16(qa0, kb0, sc);
      sc = WMMA_F16(qa1, kb1, sc);
      s[c] = sc;
    }

    // ---- online softmax over 64 keys --------------------------------------
    float mx[8];
#pragma unroll
    for (int r = 0; r < 8; ++r) {
      s[0][r] *= scale; s[1][r] *= scale; s[2][r] *= scale; s[3][r] *= scale;
      mx[r] = fmaxf(fmaxf(s[0][r], s[1][r]), fmaxf(s[2][r], s[3][r]));
    }
#pragma unroll
    for (int msk = 1; msk < 16; msk <<= 1)
#pragma unroll
      for (int r = 0; r < 8; ++r)
        mx[r] = fmaxf(mx[r], __shfl_xor(mx[r], msk, 32));

    float sum[8], alpha[8];
#pragma unroll
    for (int r = 0; r < 8; ++r) {
      const float mnew = fmaxf(mrow[r], mx[r]);
      alpha[r] = __expf(mrow[r] - mnew);
      mrow[r] = mnew;
      s[0][r] = __expf(s[0][r] - mnew);
      s[1][r] = __expf(s[1][r] - mnew);
      s[2][r] = __expf(s[2][r] - mnew);
      s[3][r] = __expf(s[3][r] - mnew);
      sum[r] = (s[0][r] + s[1][r]) + (s[2][r] + s[3][r]);
    }
#pragma unroll
    for (int msk = 1; msk < 16; msk <<= 1)
#pragma unroll
      for (int r = 0; r < 8; ++r)
        sum[r] += __shfl_xor(sum[r], msk, 32);
#pragma unroll
    for (int r = 0; r < 8; ++r) lrow[r] = lrow[r] * alpha[r] + sum[r];
#pragma unroll
    for (int f = 0; f < 4; ++f)
#pragma unroll
      for (int r = 0; r < 8; ++r) o[f][r] *= alpha[r];

    // ---- P: C-fragment layout -> A-fragment layout through LDS ------------
#pragma unroll
    for (int c = 0; c < 4; ++c)
#pragma unroll
      for (int r = 0; r < 8; ++r)
        Plds[wave][grp * 8 + r][c * 16 + nlo] = (_Float16)s[c][r];
    __syncthreads();
    const v16h pa0 = load_a_frag(&Plds[wave][0][0],      64);  // keys j0+0..31
    const v16h pa1 = load_a_frag(&Plds[wave][0][0] + 32, 64);  // keys j0+32..63

    // ---- O += P(16x64) @ V(64x64) -----------------------------------------
#pragma unroll
    for (int f = 0; f < 4; ++f) {
      const _Float16* vp = Vp + (size_t)(f * 16) * ATT_S + j0;
      v16h vb0 = load_b_frag(vp,      ATT_S);
      v16h vb1 = load_b_frag(vp + 32, ATT_S);
      o[f] = WMMA_F16(pa0, vb0, o[f]);
      o[f] = WMMA_F16(pa1, vb1, o[f]);
    }
    __syncthreads();
  }

  // ---- normalize and store as [b, s, h*64+d] f16 ---------------------------
#pragma unroll
  for (int f = 0; f < 4; ++f) {
#pragma unroll
    for (int r = 0; r < 8; ++r) {
      const int m = q0 + grp * 8 + r;
      const float val = o[f][r] / lrow[r];
      AO[((size_t)(b * ATT_S + m)) * ATT_D + h * ATT_DH + f * 16 + nlo] =
          (_Float16)val;
    }
  }
}

// ------------------------------ output GEMM --------------------------------
// out[4096,1024] = AO[4096,1024] @ Wout + b_out  (fp32 result)
__global__ __launch_bounds__(256) void out_gemm_kernel(
    const _Float16* __restrict__ AO,    // [4096,1024]
    const _Float16* __restrict__ Wt,    // [1024,1024] (N-major)
    const float*    __restrict__ bias,  // [1024]
    float* __restrict__ out) {
  const int lane = threadIdx.x & 31;
  const int wave = threadIdx.x >> 5;
  const int m0 = blockIdx.x * 128 + (wave >> 1) * 32;
  const int n0 = blockIdx.y * 128 + (wave & 1) * 64;

  const v8f zero = {0.f, 0.f, 0.f, 0.f, 0.f, 0.f, 0.f, 0.f};
  v8f acc[2][4];
#pragma unroll
  for (int im = 0; im < 2; ++im)
#pragma unroll
    for (int jn = 0; jn < 4; ++jn) acc[im][jn] = zero;

  for (int k0 = 0; k0 < ATT_D; k0 += 32) {
    v16h a0 = load_a_frag(AO + (size_t)m0        * ATT_D + k0, ATT_D);
    v16h a1 = load_a_frag(AO + (size_t)(m0 + 16) * ATT_D + k0, ATT_D);
    v16h b0 = load_b_frag(Wt + (size_t)(n0)      * ATT_D + k0, ATT_D);
    v16h b1 = load_b_frag(Wt + (size_t)(n0 + 16) * ATT_D + k0, ATT_D);
    v16h b2 = load_b_frag(Wt + (size_t)(n0 + 32) * ATT_D + k0, ATT_D);
    v16h b3 = load_b_frag(Wt + (size_t)(n0 + 48) * ATT_D + k0, ATT_D);
    acc[0][0] = WMMA_F16(a0, b0, acc[0][0]);
    acc[0][1] = WMMA_F16(a0, b1, acc[0][1]);
    acc[0][2] = WMMA_F16(a0, b2, acc[0][2]);
    acc[0][3] = WMMA_F16(a0, b3, acc[0][3]);
    acc[1][0] = WMMA_F16(a1, b0, acc[1][0]);
    acc[1][1] = WMMA_F16(a1, b1, acc[1][1]);
    acc[1][2] = WMMA_F16(a1, b2, acc[1][2]);
    acc[1][3] = WMMA_F16(a1, b3, acc[1][3]);
  }

  const int nlo = lane & 15;
  const int mb  = (lane >> 4) << 3;
#pragma unroll
  for (int im = 0; im < 2; ++im) {
#pragma unroll
    for (int jn = 0; jn < 4; ++jn) {
      const int n = n0 + jn * 16 + nlo;
      const float bv = bias[n];
#pragma unroll
      for (int r = 0; r < 8; ++r) {
        const int m = m0 + im * 16 + mb + r;
        out[(size_t)m * ATT_D + n] = acc[im][jn][r] + bv;
      }
    }
  }
}

// ------------------------------- launcher ----------------------------------
extern "C" void kernel_launch(void* const* d_in, const int* in_sizes, int n_in,
                              void* d_out, int out_size, void* d_ws, size_t ws_size,
                              hipStream_t stream) {
  const float* x     = (const float*)d_in[0];   // [2,2048,1024]
  const float* w_qkv = (const float*)d_in[1];   // [1024,3072]
  const float* b_qkv = (const float*)d_in[2];   // [3072]
  const float* w_out = (const float*)d_in[3];   // [1024,1024]
  const float* b_out = (const float*)d_in[4];   // [1024]
  float* out = (float*)d_out;

  char* ws = (char*)d_ws;
  _Float16* Xh     = (_Float16*)(ws);                 // 4096*1024
  _Float16* Wqkvt  = (_Float16*)(ws + 8388608);       // 3072*1024
  _Float16* Woutt  = (_Float16*)(ws + 14680064);      // 1024*1024
  _Float16* Qb     = (_Float16*)(ws + 16777216);      // 2*16*2048*64
  _Float16* Kb     = (_Float16*)(ws + 25165824);
  _Float16* Vtb    = (_Float16*)(ws + 33554432);
  _Float16* AO     = (_Float16*)(ws + 41943040);      // 4096*1024
  // total workspace use: 50331648 bytes

  const int nx = ATT_M * ATT_D;            // 4194304
  cvt_f32_to_f16_kernel<<<(nx + 255) / 256, 256, 0, stream>>>(x, Xh, nx);

  const int nwq = ATT_D * ATT_3D;          // 3145728
  transpose_cvt_kernel<<<(nwq + 255) / 256, 256, 0, stream>>>(w_qkv, Wqkvt, ATT_D, ATT_3D);
  const int nwo = ATT_D * ATT_D;           // 1048576
  transpose_cvt_kernel<<<(nwo + 255) / 256, 256, 0, stream>>>(w_out, Woutt, ATT_D, ATT_D);

  qkv_gemm_kernel<<<dim3(ATT_M / 128, ATT_3D / 128), 256, 0, stream>>>(
      Xh, Wqkvt, b_qkv, Qb, Kb, Vtb);

  attn_kernel<<<dim3(ATT_S / 64, ATT_B * ATT_H), 128, 0, stream>>>(
      Qb, Kb, Vtb, AO);

  out_gemm_kernel<<<dim3(ATT_M / 128, ATT_D / 128), 256, 0, stream>>>(
      AO, Woutt, b_out, out);
}
